// QFormerAttention_30296699306188
// MI455X (gfx1250) — matrix operand order repaired and checked by reading, not compiled, not run on hardware
//
#include <hip/hip_runtime.h>
#include <hip/hip_bf16.h>

typedef __attribute__((ext_vector_type(16))) _Float16 v16h;
typedef __attribute__((ext_vector_type(8)))  float    v8f;
typedef __attribute__((ext_vector_type(4)))  unsigned uint4v;  // trivially-constructible for unions

#define Hd   768
#define NHD  12
#define HDm  64
#define NQd  32
#define Bd   64
#define Pd   1024
#define Id   3072

// ---------------------------------------------------------------------------
// WMMA GEMM:  C(MxN) = A(MxK,f16) @ Bt(NxK,f16) + bias, f32 accumulate.
// One 16x64 strip per wave32: 4 N-tiles share one A fragment -> 4 WMMAs per
// 10 b128 loads. act: 0 none, 1 exact GELU.
// Cf (f32 row-major) and/or Ch (f16; trchunk==0 row-major, else block-transposed:
// row m=(b*trchunk+p), col n -> Ch[(b*N+n)*trchunk+p]).
// ---------------------------------------------------------------------------
__global__ void gemm_wmma(const _Float16* __restrict__ A, int lda,
                          const _Float16* __restrict__ Bt, int ldb,
                          const float* __restrict__ bias,
                          float* __restrict__ Cf, _Float16* __restrict__ Ch,
                          int M, int N, int K, int act, int trchunk)
{
    int wave = blockIdx.x * (blockDim.x >> 5) + (threadIdx.x >> 5);
    int lane = threadIdx.x & 31;
    int strips_n = N >> 6;
    int strips   = (M >> 4) * strips_n;
    if (wave >= strips) return;                 // wave-uniform: EXEC all-ones for WMMA
    int tm = wave / strips_n, ts = wave % strips_n;
    int hi = lane >> 4, lo = lane & 15;
    int kbA = hi * 8;                           // A frag: K = kk+kbA+j and kk+16+kbA+j
    int kbB = hi * 8;                           // B frag: K = kk+kbB+j and kk+16+kbB+j
    int n0 = ts * 64;

    const _Float16* ap  = A  + (size_t)(tm * 16 + lo) * lda;
    const _Float16* bp0 = Bt + (size_t)(n0 + lo) * ldb;
    size_t bstep = (size_t)16 * ldb;

    v8f c0, c1, c2, c3;
    float bv0 = bias ? bias[n0 + lo]      : 0.f;
    float bv1 = bias ? bias[n0 + 16 + lo] : 0.f;
    float bv2 = bias ? bias[n0 + 32 + lo] : 0.f;
    float bv3 = bias ? bias[n0 + 48 + lo] : 0.f;
#pragma unroll
    for (int r = 0; r < 8; ++r) { c0[r] = bv0; c1[r] = bv1; c2[r] = bv2; c3[r] = bv3; }

    for (int kk = 0; kk < K; kk += 32) {
        union { v16h v; uint4v q[2]; } a, b0, b1, b2, b3;
        const _Float16* aa = ap + kk + kbA;
        a.q[0] = *(const uint4v*)(aa);
        a.q[1] = *(const uint4v*)(aa + 16);
        const _Float16* bb = bp0 + kk + kbB;
        b0.q[0] = *(const uint4v*)(bb);
        b0.q[1] = *(const uint4v*)(bb + 16);
        b1.q[0] = *(const uint4v*)(bb + bstep);
        b1.q[1] = *(const uint4v*)(bb + bstep + 16);
        b2.q[0] = *(const uint4v*)(bb + 2 * bstep);
        b2.q[1] = *(const uint4v*)(bb + 2 * bstep + 16);
        b3.q[0] = *(const uint4v*)(bb + 3 * bstep);
        b3.q[1] = *(const uint4v*)(bb + 3 * bstep + 16);
        __builtin_prefetch(bb + 32, 0, 1);      // global_prefetch_b8 next K panel
        c0 = __builtin_amdgcn_wmma_f32_16x16x32_f16(false, a.v, false, b0.v, (short)0, c0, false, false);
        c1 = __builtin_amdgcn_wmma_f32_16x16x32_f16(false, a.v, false, b1.v, (short)0, c1, false, false);
        c2 = __builtin_amdgcn_wmma_f32_16x16x32_f16(false, a.v, false, b2.v, (short)0, c2, false, false);
        c3 = __builtin_amdgcn_wmma_f32_16x16x32_f16(false, a.v, false, b3.v, (short)0, c3, false, false);
    }

    auto store_tile = [&](v8f& c, int tt) {
#pragma unroll
        for (int r = 0; r < 8; ++r) {
            int m = tm * 16 + r + hi * 8;
            int col = n0 + tt * 16 + lo;
            float x = c[r];
            if (act == 1) x = 0.5f * x * (1.0f + erff(x * 0.70710678f));
            if (Cf) Cf[(size_t)m * N + col] = x;
            if (Ch) {
                if (trchunk == 0) Ch[(size_t)m * N + col] = (_Float16)x;
                else { int bb2 = m / trchunk, p = m % trchunk;
                       Ch[((size_t)bb2 * N + col) * trchunk + p] = (_Float16)x; }
            }
        }
    };
    store_tile(c0, 0); store_tile(c1, 1); store_tile(c2, 2); store_tile(c3, 3);
}

// fp32 (KxN) -> fp16 transposed (NxK)
__global__ void cvt_transpose(const float* __restrict__ src, _Float16* __restrict__ dst,
                              int K, int N)
{
    size_t i = (size_t)blockIdx.x * blockDim.x + threadIdx.x;
    if (i >= (size_t)K * N) return;
    int n = (int)(i % N), k = (int)(i / N);
    dst[(size_t)n * K + k] = (_Float16)src[i];
}

// fp32 -> fp16 straight copy (one streaming pass over vision embeddings)
__global__ void cvt_f16(const float* __restrict__ src, _Float16* __restrict__ dst, size_t n)
{
    size_t i = (size_t)blockIdx.x * blockDim.x + threadIdx.x;
    if (i < n) dst[i] = (_Float16)src[i];
}

__global__ void add_pos(const float* __restrict__ qt, const float* __restrict__ pos,
                        float* __restrict__ h, _Float16* __restrict__ h16)
{
    int i = blockIdx.x * blockDim.x + threadIdx.x;
    if (i < NQd * Hd) { float v = qt[i] + pos[i]; h[i] = v; h16[i] = (_Float16)v; }
}

// Tiny self-attention over 32 query tokens (batch-independent). 1 block/head,
// 1024 threads = 32 waves; wave == score row, lane == key index. f16 ctx out.
__global__ void self_attn(const float* __restrict__ Q, const float* __restrict__ Kk,
                          const float* __restrict__ V, _Float16* __restrict__ ctx16)
{
    __shared__ float Psm[NQd * NQd];
    int hh = blockIdx.x;
    int t  = threadIdx.x;
    int q  = t >> 5, k = t & 31;
    const float* qp = Q  + (size_t)q * Hd + hh * HDm;
    const float* kp = Kk + (size_t)k * Hd + hh * HDm;
    float s = 0.f;
    for (int d = 0; d < HDm; ++d) s += qp[d] * kp[d];
    s *= 0.125f;
    float m = s;
    for (int o = 16; o; o >>= 1) m = fmaxf(m, __shfl_xor(m, o, 32));
    float e = __expf(s - m);
    float sum = e;
    for (int o = 16; o; o >>= 1) sum += __shfl_xor(sum, o, 32);
    Psm[q * 32 + k] = e / sum;
    __syncthreads();
    float a0 = 0.f, a1 = 0.f;
    for (int kk = 0; kk < 32; ++kk) {
        float p = Psm[q * 32 + kk];
        const float* vp = V + (size_t)kk * Hd + hh * HDm + 2 * k;
        a0 += p * vp[0]; a1 += p * vp[1];
    }
    ctx16[(size_t)q * Hd + hh * HDm + 2 * k]     = (_Float16)a0;
    ctx16[(size_t)q * Hd + hh * HDm + 2 * k + 1] = (_Float16)a1;
}

// out[row] = LN(X[row] + R[row % rmod]) * g + b ; optional f16 mirror copy.
__global__ void ln_res(const float* __restrict__ X, const float* __restrict__ R, int rmod,
                       const float* __restrict__ g, const float* __restrict__ bta,
                       float* __restrict__ out, _Float16* __restrict__ out16)
{
    __shared__ float buf[Hd];
    __shared__ float red[33];
    int row = blockIdx.x, t = threadIdx.x;
    const float* xr = X + (size_t)row * Hd;
    const float* rr = R + (size_t)(row % rmod) * Hd;
    float local = 0.f;
    for (int i = t; i < Hd; i += 256) { float v = xr[i] + rr[i]; buf[i] = v; local += v; }
    for (int o = 16; o; o >>= 1) local += __shfl_xor(local, o, 32);
    if ((t & 31) == 0) red[t >> 5] = local;
    __syncthreads();
    if (t == 0) { float s = 0.f; for (int i = 0; i < 8; ++i) s += red[i]; red[32] = s * (1.0f / Hd); }
    __syncthreads();
    float mean = red[32];
    float lv = 0.f;
    for (int i = t; i < Hd; i += 256) { float d = buf[i] - mean; lv += d * d; }
    for (int o = 16; o; o >>= 1) lv += __shfl_xor(lv, o, 32);
    if ((t & 31) == 0) red[(t >> 5) + 8] = lv;
    __syncthreads();
    if (t == 0) { float s = 0.f; for (int i = 0; i < 8; ++i) s += red[i + 8];
                  red[31] = rsqrtf(s * (1.0f / Hd) + 1e-5f); }
    __syncthreads();
    float inv = red[31];
    size_t base = (size_t)row * Hd;
    for (int i = t; i < Hd; i += 256) {
        float y = (buf[i] - mean) * inv * g[i] + bta[i];
        out[base + i] = y;
        if (out16) out16[base + i] = (_Float16)y;
    }
}

// Cross attention: 1 workgroup per (batch, head), 8 waves.
// LDS: 32x1024 f32 scores (128 KB) + 32x1024 f16 probs (64 KB) = 192 KB of the
// CDNA5 WGP's 320 KB. Phase-3 A fragments are direct ds_load_b128 from f16 probs.
__global__ void cross_attn(const _Float16* __restrict__ Qc16,  // 32 x 768 f16 (batch-invariant)
                           const _Float16* __restrict__ K16,   // (B*P) x 768 f16
                           const _Float16* __restrict__ Vt16,  // B x 768 x P f16 (transposed)
                           const float* __restrict__ maskv,    // B x P f32
                           _Float16* __restrict__ ctx16)       // B x 32 x 768 f16
{
    extern __shared__ float S[];                     // 32*1024 f32
    _Float16* S16 = (_Float16*)(S + NQd * Pd);       // 32*1024 f16
    int b    = blockIdx.x / NHD;
    int hh   = blockIdx.x % NHD;
    int wave = threadIdx.x >> 5;
    int lane = threadIdx.x & 31;
    int hi   = lane >> 4, lo = lane & 15;
    int kbA  = hi * 8, kbB = hi * 8;

    // -- phase 1: S = (Q Kt) * 0.125 + mask * -1e4   (2 x 64 tiles, K=64)
    for (int tile = wave; tile < 128; tile += 8) {   // wave-uniform loop
        int tm = tile >> 6, tn = tile & 63;
        int qrow = tm * 16 + lo;
        int pcol = tn * 16 + lo;
        const _Float16* ap = Qc16 + (size_t)qrow * Hd + hh * HDm;
        const _Float16* bp = K16 + ((size_t)b * Pd + pcol) * Hd + hh * HDm;
        v8f c = {};
#pragma unroll
        for (int kk = 0; kk < HDm; kk += 32) {
            union { v16h v; uint4v q[2]; } a, bb;
            const _Float16* a0 = ap + kk + kbA;
            a.q[0] = *(const uint4v*)(a0);
            a.q[1] = *(const uint4v*)(a0 + 16);
            const _Float16* b0 = bp + kk + kbB;
            bb.q[0] = *(const uint4v*)(b0);
            bb.q[1] = *(const uint4v*)(b0 + 16);
            c = __builtin_amdgcn_wmma_f32_16x16x32_f16(false, a.v, false, bb.v, (short)0, c, false, false);
        }
        float mv = maskv[(size_t)b * Pd + pcol] * -10000.0f;
#pragma unroll
        for (int r = 0; r < 8; ++r)
            S[(tm * 16 + r + hi * 8) * Pd + pcol] = c[r] * 0.125f + mv;
    }
    __syncthreads();

    // -- phase 2: row softmax -> f16 probs (wave handles 4 rows)
    for (int r = wave * 4; r < wave * 4 + 4; ++r) {
        float mmax = -3.4e38f;
        for (int cI = lane; cI < Pd; cI += 32) mmax = fmaxf(mmax, S[r * Pd + cI]);
        for (int o = 16; o; o >>= 1) mmax = fmaxf(mmax, __shfl_xor(mmax, o, 32));
        float ssum = 0.f;
        for (int cI = lane; cI < Pd; cI += 32) {
            float e = __expf(S[r * Pd + cI] - mmax);
            S[r * Pd + cI] = e; ssum += e;
        }
        for (int o = 16; o; o >>= 1) ssum += __shfl_xor(ssum, o, 32);
        float inv = 1.0f / ssum;
        for (int cI = lane; cI < Pd; cI += 32)
            S16[r * Pd + cI] = (_Float16)(S[r * Pd + cI] * inv);
    }
    __syncthreads();

    // -- phase 3: ctx = P(32x1024,f16 in LDS) @ V(1024x64); 8 tiles, one per wave
    {
        int tm = wave >> 2, tn = wave & 3;
        int prow = tm * 16 + lo;
        int dcol = tn * 16 + lo;
        const _Float16* bp = Vt16 + ((size_t)b * Hd + hh * HDm + dcol) * Pd;
        v8f c = {};
        for (int kk = 0; kk < Pd; kk += 32) {
            union { v16h v; uint4v q[2]; } a, bb;
            const _Float16* a0 = S16 + prow * Pd + kk + kbA;
            a.q[0] = *(const uint4v*)(a0);
            a.q[1] = *(const uint4v*)(a0 + 16);
            const _Float16* b0 = bp + kk + kbB;
            bb.q[0] = *(const uint4v*)(b0);
            bb.q[1] = *(const uint4v*)(b0 + 16);
            c = __builtin_amdgcn_wmma_f32_16x16x32_f16(false, a.v, false, bb.v, (short)0, c, false, false);
        }
#pragma unroll
        for (int r = 0; r < 8; ++r) {
            int q = tm * 16 + r + hi * 8;
            ctx16[((size_t)b * NQd + q) * Hd + hh * HDm + dcol] = (_Float16)c[r];
        }
    }
}

extern "C" void kernel_launch(void* const* d_in, const int* in_sizes, int n_in,
                              void* d_out, int out_size, void* d_ws, size_t ws_size,
                              hipStream_t stream)
{
    (void)in_sizes; (void)n_in; (void)out_size; (void)ws_size;
    const float* vision  = (const float*)d_in[0];
    const float* maskv   = (const float*)d_in[1];
    const float* qtok    = (const float*)d_in[2];
    const float* pos     = (const float*)d_in[3];
    const float* sa_q_w  = (const float*)d_in[4];  const float* sa_q_b = (const float*)d_in[5];
    const float* sa_k_w  = (const float*)d_in[6];  const float* sa_k_b = (const float*)d_in[7];
    const float* sa_v_w  = (const float*)d_in[8];  const float* sa_v_b = (const float*)d_in[9];
    const float* sa_o_w  = (const float*)d_in[10]; const float* sa_o_b = (const float*)d_in[11];
    const float* sa_ln_g = (const float*)d_in[12]; const float* sa_ln_b = (const float*)d_in[13];
    const float* ca_q_w  = (const float*)d_in[14]; const float* ca_q_b = (const float*)d_in[15];
    const float* ca_k_w  = (const float*)d_in[16]; const float* ca_k_b = (const float*)d_in[17];
    const float* ca_v_w  = (const float*)d_in[18]; const float* ca_v_b = (const float*)d_in[19];
    const float* ca_o_w  = (const float*)d_in[20]; const float* ca_o_b = (const float*)d_in[21];
    const float* ca_ln_g = (const float*)d_in[22]; const float* ca_ln_b = (const float*)d_in[23];
    const float* inter_w = (const float*)d_in[24]; const float* inter_b = (const float*)d_in[25];
    const float* out_w   = (const float*)d_in[26]; const float* out_b   = (const float*)d_in[27];
    const float* ffn_ln_g= (const float*)d_in[28]; const float* ffn_ln_b= (const float*)d_in[29];

    char* ws = (char*)d_ws;
    size_t off = 0;
    auto alloc = [&](size_t bytes) -> void* {
        void* p = ws + off; off = (off + bytes + 255) & ~(size_t)255; return p;
    };

    _Float16* saq_t = (_Float16*)alloc((size_t)Hd * Hd * 2);
    _Float16* sak_t = (_Float16*)alloc((size_t)Hd * Hd * 2);
    _Float16* sav_t = (_Float16*)alloc((size_t)Hd * Hd * 2);
    _Float16* sao_t = (_Float16*)alloc((size_t)Hd * Hd * 2);
    _Float16* caq_t = (_Float16*)alloc((size_t)Hd * Hd * 2);
    _Float16* cak_t = (_Float16*)alloc((size_t)Hd * Hd * 2);
    _Float16* cav_t = (_Float16*)alloc((size_t)Hd * Hd * 2);
    _Float16* cao_t = (_Float16*)alloc((size_t)Hd * Hd * 2);
    _Float16* int_t = (_Float16*)alloc((size_t)Id * Hd * 2);          // (I x H)
    _Float16* owt_t = (_Float16*)alloc((size_t)Hd * Id * 2);          // (H x I)
    _Float16* vis16 = (_Float16*)alloc((size_t)Bd * Pd * Hd * 2);     // vision f16
    float*    hq      = (float*)alloc((size_t)NQd * Hd * 4);
    _Float16* hq16    = (_Float16*)alloc((size_t)NQd * Hd * 2);
    float*    Qs      = (float*)alloc((size_t)NQd * Hd * 4);
    float*    Ks      = (float*)alloc((size_t)NQd * Hd * 4);
    float*    Vs      = (float*)alloc((size_t)NQd * Hd * 4);
    _Float16* ctx_sa16= (_Float16*)alloc((size_t)NQd * Hd * 2);
    float*    sa_lin  = (float*)alloc((size_t)NQd * Hd * 4);
    float*    sa_out  = (float*)alloc((size_t)NQd * Hd * 4);
    _Float16* sa_out16= (_Float16*)alloc((size_t)NQd * Hd * 2);
    _Float16* Qc16    = (_Float16*)alloc((size_t)NQd * Hd * 2);
    _Float16* K16     = (_Float16*)alloc((size_t)Bd * Pd * Hd * 2);
    _Float16* Vt16    = (_Float16*)alloc((size_t)Bd * Pd * Hd * 2);
    _Float16* ctx_ca16= (_Float16*)alloc((size_t)Bd * NQd * Hd * 2);
    float*    ca_lin  = (float*)alloc((size_t)Bd * NQd * Hd * 4);
    float*    ca_out  = (float*)alloc((size_t)Bd * NQd * Hd * 4);
    _Float16* ca_out16= (_Float16*)alloc((size_t)Bd * NQd * Hd * 2);
    _Float16* ff1_16  = (_Float16*)alloc((size_t)Bd * NQd * Id * 2);
    float*    ff2     = (float*)alloc((size_t)Bd * NQd * Hd * 4);

    auto tr = [&](const float* src, _Float16* dst, int K, int N) {
        size_t tot = (size_t)K * N;
        cvt_transpose<<<(unsigned)((tot + 255) / 256), 256, 0, stream>>>(src, dst, K, N);
    };
    tr(sa_q_w, saq_t, Hd, Hd); tr(sa_k_w, sak_t, Hd, Hd);
    tr(sa_v_w, sav_t, Hd, Hd); tr(sa_o_w, sao_t, Hd, Hd);
    tr(ca_q_w, caq_t, Hd, Hd); tr(ca_k_w, cak_t, Hd, Hd);
    tr(ca_v_w, cav_t, Hd, Hd); tr(ca_o_w, cao_t, Hd, Hd);
    tr(inter_w, int_t, Hd, Id);            // -> I x H
    tr(out_w,   owt_t, Id, Hd);            // -> H x I

    {   // vision fp32 -> fp16, single streaming pass
        size_t n = (size_t)Bd * Pd * Hd;
        cvt_f16<<<(unsigned)((n + 255) / 256), 256, 0, stream>>>(vision, vis16, n);
    }

    auto gemm = [&](const _Float16* A, int lda, const _Float16* Bt, int ldb, const float* bias,
                    float* Cf, _Float16* Ch, int M, int N, int K, int act, int trchunk) {
        int strips = (M / 16) * (N / 64);
        int blocks = (strips + 3) / 4;
        gemm_wmma<<<blocks, 128, 0, stream>>>(A, lda, Bt, ldb, bias, Cf, Ch, M, N, K, act, trchunk);
    };

    // --- self attention (batch invariant) ---
    add_pos<<<(NQd * Hd + 255) / 256, 256, 0, stream>>>(qtok, pos, hq, hq16);
    gemm(hq16, Hd, saq_t, Hd, sa_q_b, Qs, nullptr, NQd, Hd, Hd, 0, 0);
    gemm(hq16, Hd, sak_t, Hd, sa_k_b, Ks, nullptr, NQd, Hd, Hd, 0, 0);
    gemm(hq16, Hd, sav_t, Hd, sa_v_b, Vs, nullptr, NQd, Hd, Hd, 0, 0);
    self_attn<<<NHD, 1024, 0, stream>>>(Qs, Ks, Vs, ctx_sa16);
    gemm(ctx_sa16, Hd, sao_t, Hd, sa_o_b, sa_lin, nullptr, NQd, Hd, Hd, 0, 0);
    ln_res<<<NQd, 256, 0, stream>>>(sa_lin, hq, NQd, sa_ln_g, sa_ln_b, sa_out, sa_out16);

    // --- cross attention ---
    gemm(sa_out16, Hd, caq_t, Hd, ca_q_b, nullptr, Qc16, NQd, Hd, Hd, 0, 0);
    gemm(vis16, Hd, cak_t, Hd, ca_k_b, nullptr, K16,  Bd * Pd, Hd, Hd, 0, 0);   // f16 row-major
    gemm(vis16, Hd, cav_t, Hd, ca_v_b, nullptr, Vt16, Bd * Pd, Hd, Hd, 0, Pd);  // f16 transposed/batch
    (void)hipFuncSetAttribute((const void*)cross_attn,
                              hipFuncAttributeMaxDynamicSharedMemorySize,
                              NQd * Pd * 4 + NQd * Pd * 2);
    cross_attn<<<Bd * NHD, 256, NQd * Pd * 4 + NQd * Pd * 2, stream>>>(
        Qc16, K16, Vt16, maskv, ctx_ca16);
    gemm(ctx_ca16, Hd, cao_t, Hd, ca_o_b, ca_lin, nullptr, Bd * NQd, Hd, Hd, 0, 0);
    ln_res<<<Bd * NQd, 256, 0, stream>>>(ca_lin, sa_out, NQd, ca_ln_g, ca_ln_b, ca_out, ca_out16);

    // --- FFN ---
    gemm(ca_out16, Hd, int_t, Hd, inter_b, nullptr, ff1_16, Bd * NQd, Id, Hd, 1, 0); // GELU fused
    gemm(ff1_16, Id, owt_t, Id, out_b, ff2, nullptr, Bd * NQd, Hd, Id, 0, 0);
    ln_res<<<Bd * NQd, 256, 0, stream>>>(ff2, ca_out, Bd * NQd, ffn_ln_g, ffn_ln_b,
                                         (float*)d_out, nullptr);
}
